// GQASelfAttention_65515431133509
// MI455X (gfx1250) — matrix operand order repaired and checked by reading, not compiled, not run on hardware
//
#include <hip/hip_runtime.h>
#include <math.h>

typedef __attribute__((ext_vector_type(16))) __bf16 v16bf;
typedef __attribute__((ext_vector_type(8)))  __bf16 v8bf;
typedef __attribute__((ext_vector_type(8)))  float  v8f;
typedef unsigned short u16;

#define B_    2
#define T_    1024
#define D_    4096
#define H_    32
#define KVH_  8
#define HD_   128
#define GRP_  4

// gfx1250 async-to-LDS path (ASYNCcnt-tracked DMA), guarded for toolchain.
#if defined(__AMDGCN__) && __has_builtin(__builtin_amdgcn_global_load_async_to_lds_b128) && __has_builtin(__builtin_amdgcn_s_wait_asynccnt)
#define USE_ASYNC_GEMM 1
#else
#define USE_ASYNC_GEMM 0
#endif

// Exact builtin parameter types (from compiler diagnostic): int4 vector
// pointers in global (1) / LDS (3) address spaces.
typedef int v4i_ __attribute__((vector_size(16)));
typedef __attribute__((address_space(1))) v4i_ g_v4i;
typedef __attribute__((address_space(3))) v4i_ l_v4i;

__device__ __forceinline__ u16 f32_to_bf16(float f) {
  unsigned u = __float_as_uint(f);
  u += 0x7fffu + ((u >> 16) & 1u);           // round-to-nearest-even
  return (u16)(u >> 16);
}

// A-fragment (16x32 MxK, bf16): lane = row (l&15); lanes 0-15 hold K 0..15,
// lanes 16-31 hold K 16..31 -> 16 contiguous elements (two 16B loads at +0,+8).
__device__ __forceinline__ v16bf frag_a(const u16* p) {
  v8bf lo = *reinterpret_cast<const v8bf*>(p);
  v8bf hi = *reinterpret_cast<const v8bf*>(p + 8);
  v16bf r;
#pragma unroll
  for (int i = 0; i < 8; ++i) { r[i] = lo[i]; r[i + 8] = hi[i]; }
  return r;
}
// B-fragment (32x16 KxN, bf16): lane = col (l&15); lanes 0-15 hold K 0..15,
// lanes 16-31 hold K 16..31 -> 16 contiguous elements (two 16B loads at +0,+8).
__device__ __forceinline__ v16bf frag_b(const u16* p) {
  v8bf lo = *reinterpret_cast<const v8bf*>(p);
  v8bf hi = *reinterpret_cast<const v8bf*>(p + 8);
  v16bf r;
#pragma unroll
  for (int i = 0; i < 8; ++i) { r[i] = lo[i]; r[i + 8] = hi[i]; }
  return r;
}
__device__ __forceinline__ v8f wmma_bf16(v16bf a, v16bf b, v8f c) {
  return __builtin_amdgcn_wmma_f32_16x16x32_bf16(false, a, false, b, (short)0, c,
                                                 false, false);
}
__device__ __forceinline__ v8f zero8() {
  v8f z;
#pragma unroll
  for (int i = 0; i < 8; ++i) z[i] = 0.0f;
  return z;
}

// ---------------------------------------------------------------- converts ---
__global__ void cvt_bf16_kernel(const float* __restrict__ src,
                                u16* __restrict__ dst, int n) {
  int i = blockIdx.x * blockDim.x + threadIdx.x;
  if (i < n) dst[i] = f32_to_bf16(src[i]);
}

// w (K x N, f32, row-major) -> wT (N x K, bf16)
__global__ void transpose_cvt_kernel(const float* __restrict__ w,
                                     u16* __restrict__ wT, int K, int N) {
  __shared__ float tile[32][33];
  int n0 = blockIdx.x * 32, k0 = blockIdx.y * 32;
  int tx = threadIdx.x, ty = threadIdx.y;          // 32 x 8
#pragma unroll
  for (int j = 0; j < 32; j += 8)
    tile[ty + j][tx] = w[(size_t)(k0 + ty + j) * N + n0 + tx];
  __syncthreads();
#pragma unroll
  for (int j = 0; j < 32; j += 8)
    wT[(size_t)(n0 + ty + j) * K + k0 + tx] = f32_to_bf16(tile[tx][ty + j]);
}

// ------------------------------------------------------------------- GEMM ----
// C(MxN,f32) = A(MxK,bf16 row-major) * Bt(NxK,bf16)^T.  Block tile 128x128,
// 8 waves (2x4), wave tile 64x32 = 4x2 subtiles of 16x16.
// Async variant: K staged in 64-wide double-buffered LDS tiles filled by
// GLOBAL_LOAD_ASYNC_TO_LDS_B128 (ASYNCcnt), WMMA consumes via ds_load_b128.
#define APITCH      72                       // padded row pitch (bank spread)
#define TILE_ELEMS  (128 * APITCH)           // 9216 elements per tile
#define STAGE_ELEMS (2 * TILE_ELEMS)         // A tile + B tile
#define GEMM_SMEM   (2 * STAGE_ELEMS * 2)    // double buffer, bytes = 73728

__global__ __launch_bounds__(256) void gemm_bf16_kernel(
    const u16* __restrict__ A, const u16* __restrict__ Bt,
    float* __restrict__ C, int M, int N, int K) {
  const int l   = threadIdx.x & 31;
  const int wid = threadIdx.x >> 5;
  const int mw  = wid >> 2;                        // 0..1
  const int nw  = wid & 3;                         // 0..3
  const int mr0 = blockIdx.y * 128;
  const int nc0 = blockIdx.x * 128;
  const int lr  = l & 15;
  const int hl  = l >> 4;

  v8f acc[4][2];
#pragma unroll
  for (int mi = 0; mi < 4; ++mi)
#pragma unroll
    for (int ni = 0; ni < 2; ++ni) acc[mi][ni] = zero8();

#if USE_ASYNC_GEMM
  extern __shared__ char gsm[];
  u16* sm = reinterpret_cast<u16*>(gsm);
  const int r4 = l >> 3;                           // 0..3 rows per b128 op
  const int c8 = (l & 7) * 8;                      // 8-elem (16B) column chunk

  // Each wave DMA-fills 16 rows of A and 16 rows of Bt per stage:
  // 4 async b128 ops each cover 4 rows x 64 cols (32 lanes x 16B).
  auto fill = [&](int st, int kb) {
    u16* aB = sm + st * STAGE_ELEMS;
    u16* bB = aB + TILE_ELEMS;
    const u16* Ag = A  + (size_t)(mr0 + wid * 16 + r4) * K + kb + c8;
    const u16* Bg = Bt + (size_t)(nc0 + wid * 16 + r4) * K + kb + c8;
#pragma unroll
    for (int i = 0; i < 4; ++i) {
      __builtin_amdgcn_global_load_async_to_lds_b128(
          (g_v4i*)(Ag + (size_t)i * 4 * K),
          (l_v4i*)(aB + (wid * 16 + i * 4 + r4) * APITCH + c8), 0, 0);
      __builtin_amdgcn_global_load_async_to_lds_b128(
          (g_v4i*)(Bg + (size_t)i * 4 * K),
          (l_v4i*)(bB + (wid * 16 + i * 4 + r4) * APITCH + c8), 0, 0);
    }
  };

  fill(0, 0);
  __builtin_amdgcn_s_wait_asynccnt(0);
  __syncthreads();

  int st = 0;
  for (int kb = 0; kb < K; kb += 64) {
    if (kb + 64 < K) fill(st ^ 1, kb + 64);        // DMA next stage, overlap
    const u16* aB = sm + st * STAGE_ELEMS;
    const u16* bB = aB + TILE_ELEMS;
#pragma unroll
    for (int kk = 0; kk < 64; kk += 32) {
      v16bf af[4], bfr[2];
#pragma unroll
      for (int mi = 0; mi < 4; ++mi)
        af[mi] = frag_a(aB + (mw * 64 + mi * 16 + lr) * APITCH + kk + hl * 16);
#pragma unroll
      for (int ni = 0; ni < 2; ++ni)
        bfr[ni] = frag_b(bB + (nw * 32 + ni * 16 + lr) * APITCH + kk + hl * 16);
#pragma unroll
      for (int mi = 0; mi < 4; ++mi)
#pragma unroll
        for (int ni = 0; ni < 2; ++ni)
          acc[mi][ni] = wmma_bf16(af[mi], bfr[ni], acc[mi][ni]);
    }
    __builtin_amdgcn_s_wait_asynccnt(0);
    __syncthreads();
    st ^= 1;
  }
#else
  // Fallback: direct-global WMMA loop (L2-resident operands).
  const u16* Ap = A  + (size_t)(mr0 + mw * 64 + lr) * K + hl * 16;
  const u16* Bp = Bt + (size_t)(nc0 + nw * 32 + lr) * K + hl * 16;
  for (int kb = 0; kb < K; kb += 32) {
    v16bf af[4], bfr[2];
#pragma unroll
    for (int mi = 0; mi < 4; ++mi) af[mi] = frag_a(Ap + (size_t)mi * 16 * K + kb);
#pragma unroll
    for (int ni = 0; ni < 2; ++ni) bfr[ni] = frag_b(Bp + (size_t)ni * 16 * K + kb);
#pragma unroll
    for (int mi = 0; mi < 4; ++mi)
#pragma unroll
      for (int ni = 0; ni < 2; ++ni)
        acc[mi][ni] = wmma_bf16(af[mi], bfr[ni], acc[mi][ni]);
  }
#endif

#pragma unroll
  for (int mi = 0; mi < 4; ++mi)
#pragma unroll
    for (int ni = 0; ni < 2; ++ni)
#pragma unroll
      for (int r = 0; r < 8; ++r) {
        int row = mr0 + mw * 64 + mi * 16 + r + 8 * hl;  // C: M = r + 8*(l>=16)
        int col = nc0 + nw * 32 + ni * 16 + lr;          //    N = l & 15
        C[(size_t)row * N + col] = acc[mi][ni][r];
      }
}

// -------------------------------------------------------------- RoPE pack ----
// src: (b,t,nh,hd) f32  ->  dst: (b,nh,t,hd) bf16 with interleaved RoPE
__global__ void rope_pack_kernel(const float* __restrict__ src,
                                 u16* __restrict__ dst, int nh,
                                 const int* __restrict__ sp) {
  int idx = blockIdx.x * blockDim.x + threadIdx.x;
  int total = B_ * T_ * nh * HD_;
  if (idx >= total) return;
  int d    = idx & (HD_ - 1);
  int rest = idx >> 7;
  int hh   = rest % nh;  rest /= nh;
  int t    = rest & (T_ - 1);
  int b    = rest >> 10;
  float xv = src[idx];
  float pv = src[idx ^ 1];                          // pair element
  float pos = (float)(t + sp[0]);
  int   j   = d & 63;                               // emb index mod 64
  float ang = pos * powf(10000.0f, -(float)j * (1.0f / 64.0f));
  float c = cosf(ang), sn = sinf(ang);
  float o = (d & 1) ? (xv * c + pv * sn) : (xv * c - pv * sn);
  dst[((size_t)((b * nh + hh) * T_ + t)) * HD_ + d] = f32_to_bf16(o);
}

// src: (b,t,kvh,hd) f32 -> dst: (b,kvh,hd,t) bf16 (transposed for PV B-frags)
__global__ void v_pack_kernel(const float* __restrict__ src,
                              u16* __restrict__ dst) {
  int idx = blockIdx.x * blockDim.x + threadIdx.x;
  int total = B_ * T_ * KVH_ * HD_;
  if (idx >= total) return;
  int d    = idx & (HD_ - 1);
  int rest = idx >> 7;
  int kvh  = rest & (KVH_ - 1);  rest >>= 3;
  int t    = rest & (T_ - 1);
  int b    = rest >> 10;
  dst[((size_t)((b * KVH_ + kvh) * HD_ + d)) * T_ + t] = f32_to_bf16(src[idx]);
}

// -------------------------------------------------------------- attention ----
// One workgroup per (b, h, 16-query tile).  8 waves.
// Phase 1: S = scale * q K^T (causal tiles only) -> LDS (f32)
// Phase 2: row softmax (16 threads/row, shfl width 16) -> LDS P (bf16)
// Phase 3: O = P V via WMMA from LDS A-frags, write (b,t,h*hd) bf16.
#define SS_ 1028   // f32 score row stride (padded, not mult of 8 -> bank spread)
#define PS_ 1032   // bf16 prob row stride (mult of 8 -> 16B-aligned frag loads)
#define ATTN_SMEM (16 * SS_ * 4 + 16 * PS_ * 2)

__global__ __launch_bounds__(256) void attn_kernel(
    const u16* __restrict__ qb, const u16* __restrict__ kb,
    const u16* __restrict__ vt, u16* __restrict__ ob) {
  extern __shared__ char smem[];
  float* s = reinterpret_cast<float*>(smem);
  u16*   p = reinterpret_cast<u16*>(smem + 16 * SS_ * 4);

  const int bx  = blockIdx.x;                 // query tile (16 rows)
  const int h   = blockIdx.y;
  const int b   = blockIdx.z;
  const int kvh = h / GRP_;
  const int tid = threadIdx.x;
  const int l   = tid & 31;
  const int wid = tid >> 5;
  const int lr  = l & 15;
  const int hl  = l >> 4;
  const int qr0 = bx * 16;
  const int nk  = (bx + 1) * 16;              // valid key columns
  const int nkR = (nk + 31) & ~31;            // rounded for K=32 WMMA steps

  const u16* qbase = qb + (size_t)(b * H_ + h)     * T_ * HD_;
  const u16* kbase = kb + (size_t)(b * KVH_ + kvh) * T_ * HD_;
  const u16* vbase = vt + (size_t)(b * KVH_ + kvh) * HD_ * T_;

  // Q fragments for this 16-row tile (HD=128 -> 4 K-chunks), kept in VGPRs.
  v16bf aq[4];
  {
    const u16* qp = qbase + (size_t)(qr0 + lr) * HD_ + hl * 16;
#pragma unroll
    for (int kk = 0; kk < 4; ++kk) aq[kk] = frag_a(qp + kk * 32);
  }
  const float scale = 0.08838834764831845f;   // 1/sqrt(128)

  // ---- Phase 1: scores (causal key tiles split across 8 waves) ----
  for (int kt = wid; kt <= bx; kt += 8) {
    v8f acc = zero8();
    const u16* kp = kbase + (size_t)(kt * 16 + lr) * HD_ + hl * 16;
#pragma unroll
    for (int kk = 0; kk < 4; ++kk)
      acc = wmma_bf16(aq[kk], frag_b(kp + kk * 32), acc);
#pragma unroll
    for (int r = 0; r < 8; ++r) {
      int row = r + 8 * hl;
      int qg  = qr0 + row;
      int kg  = kt * 16 + lr;
      float v = acc[r] * scale;
      if (kg > qg) v = -1e30f;                // causal mask
      s[row * SS_ + kt * 16 + lr] = v;
    }
  }
  __syncthreads();

  // ---- Phase 2: softmax, 16 threads per row ----
  {
    int row = tid >> 4;
    int sub = tid & 15;
    float m = -1e30f;
    for (int j = sub; j < nk; j += 16) m = fmaxf(m, s[row * SS_ + j]);
#pragma unroll
    for (int off = 8; off; off >>= 1) m = fmaxf(m, __shfl_xor(m, off, 16));
    float sum = 0.0f;
    for (int j = sub; j < nk; j += 16) sum += __expf(s[row * SS_ + j] - m);
#pragma unroll
    for (int off = 8; off; off >>= 1) sum += __shfl_xor(sum, off, 16);
    float inv = 1.0f / sum;
    for (int j = sub; j < nk; j += 16)
      p[row * PS_ + j] = f32_to_bf16(__expf(s[row * SS_ + j] - m) * inv);
    if (nkR > nk) p[row * PS_ + nk + sub] = 0;  // zero pad partial K-step
  }
  __syncthreads();

  // ---- Phase 3: O = P @ V, wave wid owns head-dim tile [wid*16, wid*16+16) ----
  {
    const int dt0 = wid * 16;
    v8f acc = zero8();
    for (int kb2 = 0; kb2 < nkR; kb2 += 32) {
      v16bf ap = frag_a(p + (size_t)lr * PS_ + kb2 + hl * 16);
      v16bf bv = frag_b(vbase + (size_t)(dt0 + lr) * T_ + kb2 + hl * 16);
      acc = wmma_bf16(ap, bv, acc);
    }
#pragma unroll
    for (int r = 0; r < 8; ++r) {
      int row = r + 8 * hl;
      int t   = qr0 + row;
      int d   = dt0 + lr;
      ob[(size_t)(b * T_ + t) * (H_ * HD_) + h * HD_ + d] = f32_to_bf16(acc[r]);
    }
  }
}

// ---------------------------------------------------------------- launch -----
extern "C" void kernel_launch(void* const* d_in, const int* in_sizes, int n_in,
                              void* d_out, int out_size, void* d_ws,
                              size_t ws_size, hipStream_t stream) {
  (void)in_sizes; (void)n_in; (void)out_size; (void)ws_size;
  const float* x  = (const float*)d_in[0];
  const float* wq = (const float*)d_in[1];
  const float* wk = (const float*)d_in[2];
  const float* wv = (const float*)d_in[3];
  const float* wo = (const float*)d_in[4];
  const int*   sp = (const int*)d_in[5];

  char* ws = (char*)d_ws;
  const size_t M   = (size_t)B_ * T_;           // 2048
  const size_t SZQ = M * D_;                    // 2048*4096
  const size_t SZK = M * (KVH_ * HD_);          // 2048*1024
  // workspace layout (bf16 buffers unless noted)
  u16* xb  = (u16*)(ws);                                    // 16 MB
  u16* wqT = (u16*)(ws + 16u  * 1024 * 1024);               // 32 MB
  u16* wkT = (u16*)(ws + 48u  * 1024 * 1024);               //  8 MB
  u16* wvT = (u16*)(ws + 56u  * 1024 * 1024);               //  8 MB
  u16* woT = (u16*)(ws + 64u  * 1024 * 1024);               // 32 MB
  float* qf = (float*)(ws + 96u  * 1024 * 1024);            // 32 MB f32
  float* kf = (float*)(ws + 128u * 1024 * 1024);            //  8 MB f32
  float* vf = (float*)(ws + 136u * 1024 * 1024);            //  8 MB f32
  u16* qbp = (u16*)(ws + 144u * 1024 * 1024);               // 16 MB
  u16* kbp = (u16*)(ws + 160u * 1024 * 1024);               //  4 MB
  u16* vtp = (u16*)(ws + 164u * 1024 * 1024);               //  4 MB
  u16* obp = (u16*)(ws + 96u  * 1024 * 1024);               // reuse qf (16 MB)

  // 1) convert activations + transpose-convert weights to bf16
  cvt_bf16_kernel<<<(int)(SZQ / 256), 256, 0, stream>>>(x, xb, (int)SZQ);
  transpose_cvt_kernel<<<dim3(128, 128), dim3(32, 8), 0, stream>>>(wq, wqT, D_, 4096);
  transpose_cvt_kernel<<<dim3(32, 128),  dim3(32, 8), 0, stream>>>(wk, wkT, D_, 1024);
  transpose_cvt_kernel<<<dim3(32, 128),  dim3(32, 8), 0, stream>>>(wv, wvT, D_, 1024);
  transpose_cvt_kernel<<<dim3(128, 128), dim3(32, 8), 0, stream>>>(wo, woT, 4096, D_);

  // 2) QKV projections (WMMA bf16 -> f32), async-LDS double-buffered
  gemm_bf16_kernel<<<dim3(32, 16), 256, GEMM_SMEM, stream>>>(xb, wqT, qf, (int)M, 4096, D_);
  gemm_bf16_kernel<<<dim3(8, 16),  256, GEMM_SMEM, stream>>>(xb, wkT, kf, (int)M, 1024, D_);
  gemm_bf16_kernel<<<dim3(8, 16),  256, GEMM_SMEM, stream>>>(xb, wvT, vf, (int)M, 1024, D_);

  // 3) RoPE + layout packs
  rope_pack_kernel<<<(int)(SZQ / 256), 256, 0, stream>>>(qf, qbp, H_, sp);
  rope_pack_kernel<<<(int)(SZK / 256), 256, 0, stream>>>(kf, kbp, KVH_, sp);
  v_pack_kernel<<<(int)(SZK / 256), 256, 0, stream>>>(vf, vtp);

  // 4) causal GQA attention (WMMA QK^T + WMMA PV)
  attn_kernel<<<dim3(T_ / 16, H_, B_), 256, ATTN_SMEM, stream>>>(qbp, kbp, vtp, obp);

  // 5) output projection -> f32 d_out
  gemm_bf16_kernel<<<dim3(32, 16), 256, GEMM_SMEM, stream>>>(obp, woT, (float*)d_out,
                                                             (int)M, 4096, D_);
}
